// DCNV4_YOLO_16140487098591
// MI455X (gfx1250) — matrix operand order repaired and checked by reading, not compiled, not run on hardware
//
#include <hip/hip_runtime.h>
#include <hip/hip_bf16.h>

typedef __attribute__((ext_vector_type(16))) _Float16 v16h;
typedef __attribute__((ext_vector_type(8)))  _Float16 v8h;
typedef __attribute__((ext_vector_type(4)))  _Float16 v4h;
typedef __attribute__((ext_vector_type(8)))  float    v8f;

#define MTOT   16384   // N*H*W
#define CCH    256
#define HW     4096
#define LDA    48      // LDS row stride in halves (96B; 16/32B-aligned chunks)
#define BM     128
#define BN     64
#define BK     32

// ---------------------------------------------------------------------------
// CDNA5 async global -> LDS copy (ASYNCcnt path), 16B per lane
// ---------------------------------------------------------------------------
__device__ __forceinline__ void async_cp16(unsigned lds_off, const _Float16* g)
{
    asm volatile("global_load_async_to_lds_b128 %0, %1, off"
                 :: "v"(lds_off), "v"(g) : "memory");
}
__device__ __forceinline__ void wait_async0()
{
    asm volatile("s_wait_asynccnt 0x0" ::: "memory");
}

// ---------------------------------------------------------------------------
// x (NCHW fp32) -> xt16 (M x 256, f16).  32x32 LDS transpose tiles.
// ---------------------------------------------------------------------------
__global__ void __launch_bounds__(256) convert_x_kernel(
    const float* __restrict__ x, _Float16* __restrict__ xt)
{
    __shared__ float tile[32][33];
    int t  = blockIdx.x;
    int n  = t >> 10;
    int r  = t & 1023;
    int ct = r >> 7;            // 0..7   channel tile
    int pt = r & 127;           // 0..127 pixel tile
    int c0 = ct * 32, hw0 = pt * 32;
    int col  = threadIdx.x & 31;
    int row8 = threadIdx.x >> 5;
    #pragma unroll
    for (int rr = row8; rr < 32; rr += 8)
        tile[rr][col] = x[((size_t)(n * CCH + c0 + rr) << 12) + hw0 + col];
    __syncthreads();
    #pragma unroll
    for (int rr = row8; rr < 32; rr += 8)
        xt[(size_t)(n * HW + hw0 + rr) * CCH + c0 + col] = (_Float16)tile[col][rr];
}

// ---------------------------------------------------------------------------
// fp32 weight [rows_src][256] -> f16 [rows_dst][256] with zero padding
// ---------------------------------------------------------------------------
__global__ void convert_w_kernel(const float* __restrict__ src,
                                 _Float16* __restrict__ dst,
                                 int rows_src, int rows_dst)
{
    int idx = blockIdx.x * 256 + threadIdx.x;
    if (idx >= rows_dst * CCH) return;
    int r = idx >> 8;
    dst[idx] = (r < rows_src) ? (_Float16)src[idx] : (_Float16)0.0f;
}

// ---------------------------------------------------------------------------
// WMMA GEMM:  out[M][n] = A(f16)[M][256] @ B(f16)[n][256]^T + bias
//   tile 128x64, BK=32, 8 waves; wave = 16-row strip x 64 cols (4 WMMA/step)
//   double-buffered LDS fed by global_load_async_to_lds_b128.
//   Loader: thread t copies A chunks (row, row+64) and B chunk (row) where
//   row = t>>2, seg = (t&3)*8 -- branch-free, 3 async issues per K-step.
// mode 0: fp32 store (stride ldc, guarded n<nreal)
// mode 1: fused BN(inference) + SiLU, transposed NCHW store
// ---------------------------------------------------------------------------
__global__ void __launch_bounds__(256) gemm_wmma_kernel(
    const _Float16* __restrict__ A,
    const _Float16* __restrict__ B,
    const float* __restrict__ bias,
    float* __restrict__ out,
    int nreal, int ldc, int mode,
    const float* __restrict__ gamma, const float* __restrict__ beta,
    const float* __restrict__ rmean, const float* __restrict__ rvar)
{
    __shared__ _Float16 As[2][BM * LDA];   // 2 x 12 KB
    __shared__ _Float16 Bs[2][BN * LDA];   // 2 x  6 KB

    const int tid  = threadIdx.x;
    const int gm   = blockIdx.x * BM;
    const int gn   = blockIdx.y * BN;
    const int lane = tid & 31;
    const int wave = tid >> 5;       // 0..7 -> 16-row strip
    const int lhi  = lane >> 4;      // K-half selector (contiguous k = 16*lhi + s in A and B)
    const int l16  = lane & 15;

    // ---- hoisted loader coordinates (branch-free) ----
    const int lrow = tid >> 2;            // 0..63
    const int lseg = (tid & 3) * 8;       // 0,8,16,24 halves

    const _Float16* aG0 = A + (size_t)(gm + lrow) * CCH + lseg;
    const _Float16* aG1 = aG0 + (size_t)64 * CCH;
    const _Float16* bG  = B + (size_t)(gn + lrow) * CCH + lseg;

    unsigned aL0[2], aL1[2], bL[2];
    #pragma unroll
    for (int s = 0; s < 2; ++s) {
        aL0[s] = (unsigned)(uintptr_t)&As[s][lrow * LDA + lseg];
        aL1[s] = (unsigned)(uintptr_t)&As[s][(lrow + 64) * LDA + lseg];
        bL[s]  = (unsigned)(uintptr_t)&Bs[s][lrow * LDA + lseg];
    }

    v8f acc[4] = {{}, {}, {}, {}};

    // prologue: stage tile 0
    async_cp16(aL0[0], aG0);
    async_cp16(aL1[0], aG1);
    async_cp16(bL[0],  bG);
    wait_async0();
    __syncthreads();

    int stage = 0;
    #pragma unroll
    for (int k0 = 0; k0 < 256; k0 += BK) {
        if (k0 + BK < 256) {                 // overlap next-tile DMA w/ WMMA
            int nxt = stage ^ 1;
            async_cp16(aL0[nxt], aG0 + k0 + BK);
            async_cp16(aL1[nxt], aG1 + k0 + BK);
            async_cp16(bL[nxt],  bG  + k0 + BK);
        }

        const _Float16* Ab = &As[stage][0];
        const _Float16* Bb = &Bs[stage][0];

        // A fragment: row = M (lane&15), lane-hi selects K subsets
        int ar = wave * 16 + l16;
        v8h a0 = *(const v8h*)&Ab[ar * LDA + lhi * 16];       // K 0..7  / 16..23
        v8h a1 = *(const v8h*)&Ab[ar * LDA + lhi * 16 + 8];   // K 8..15 / 24..31
        v16h af = __builtin_shufflevector(a0, a1,
                    0,1,2,3,4,5,6,7,8,9,10,11,12,13,14,15);

        #pragma unroll
        for (int j = 0; j < 4; ++j) {
            v16h bf = *(const v16h*)&Bb[(j * 16 + l16) * LDA + lhi * 16];
            acc[j] = __builtin_amdgcn_wmma_f32_16x16x32_f16(
                         false, af, false, bf, (short)0, acc[j], false, false);
        }

        wait_async0();      // next tile landed
        __syncthreads();    // all waves done reading current tile
        stage ^= 1;
    }

    // C/D layout: VGPR r -> M = wave*16 + 8*lhi + r, N = gn + j*16 + (lane&15)
    #pragma unroll
    for (int j = 0; j < 4; ++j) {
        int n = gn + j * 16 + l16;
        #pragma unroll
        for (int r = 0; r < 8; ++r) {
            int mrow = gm + wave * 16 + lhi * 8 + r;
            float y = acc[j][r];
            if (mode == 0) {
                if (n < nreal) out[(size_t)mrow * ldc + n] = y + bias[n];
            } else {
                int nb = mrow >> 12, hw = mrow & 4095;
                y += bias[n];
                float inv = gamma[n] * rsqrtf(rvar[n] + 1e-5f);
                y = (y - rmean[n]) * inv + beta[n];
                float s = 1.0f / (1.0f + __expf(-y));
                out[((size_t)(nb * CCH + n) << 12) + hw] = y * s;
            }
        }
    }
}

// ---------------------------------------------------------------------------
// Deformable bilinear sampling + mask reduce.
// thread = (pixel, group, 4-channel slice).  v: [M][256] fp32. om: [M][112].
// ---------------------------------------------------------------------------
__global__ void __launch_bounds__(256) sample_kernel(
    const float* __restrict__ v, const float* __restrict__ om,
    _Float16* __restrict__ samp)
{
    int t      = blockIdx.x * 256 + threadIdx.x;
    int lane16 = t & 15;
    int pg     = t >> 4;
    int g      = pg & 3;
    int p      = pg >> 2;
    int n      = p >> 12;
    int hw     = p & 4095;
    int h      = hw >> 6, w = hw & 63;

    const float* omg = om + (size_t)p * 112 + g * 27;
    const float* vb  = v + ((size_t)n << 12) * CCH;
    int cb = g * 64 + lane16 * 4;

    float acc0 = 0.f, acc1 = 0.f, acc2 = 0.f, acc3 = 0.f;

    #pragma unroll
    for (int k = 0; k < 9; ++k) {
        float ox = omg[2 * k];
        float oy = omg[2 * k + 1];
        float mv = omg[18 + k];
        float lh = (float)(h - 1 + (k / 3)) + oy;
        float lw = (float)(w - 1 + (k % 3)) + ox;
        float h0f = floorf(lh), w0f = floorf(lw);
        float fh = lh - h0f, fw = lw - w0f;
        int h0 = (int)h0f, w0 = (int)w0f;
        float wgt[4] = { (1.f - fh) * (1.f - fw), (1.f - fh) * fw,
                         fh * (1.f - fw),          fh * fw };
        int   dh[4]  = { 0, 0, 1, 1 };
        int   dw[4]  = { 0, 1, 0, 1 };
        float s0 = 0.f, s1 = 0.f, s2 = 0.f, s3 = 0.f;
        #pragma unroll
        for (int c = 0; c < 4; ++c) {
            int hi = h0 + dh[c], wi = w0 + dw[c];
            bool valid = (hi >= 0) & (hi < 64) & (wi >= 0) & (wi < 64);
            int hc = hi < 0 ? 0 : (hi > 63 ? 63 : hi);
            int wc = wi < 0 ? 0 : (wi > 63 ? 63 : wi);
            float4 val = *(const float4*)(vb + ((size_t)(hc * 64 + wc)) * CCH + cb);
            float ww = valid ? wgt[c] : 0.f;
            s0 += ww * val.x; s1 += ww * val.y; s2 += ww * val.z; s3 += ww * val.w;
        }
        acc0 += mv * s0; acc1 += mv * s1; acc2 += mv * s2; acc3 += mv * s3;
    }

    v4h o;
    o[0] = (_Float16)acc0; o[1] = (_Float16)acc1;
    o[2] = (_Float16)acc2; o[3] = (_Float16)acc3;
    *(v4h*)(samp + (size_t)p * CCH + cb) = o;
}

// ---------------------------------------------------------------------------
extern "C" void kernel_launch(void* const* d_in, const int* in_sizes, int n_in,
                              void* d_out, int out_size, void* d_ws, size_t ws_size,
                              hipStream_t stream)
{
    const float* x     = (const float*)d_in[0];
    const float* Wv    = (const float*)d_in[1];
    const float* bv    = (const float*)d_in[2];
    const float* Wom   = (const float*)d_in[3];
    const float* bom   = (const float*)d_in[4];
    const float* Wo    = (const float*)d_in[5];
    const float* bo    = (const float*)d_in[6];
    const float* gamma = (const float*)d_in[7];
    const float* beta  = (const float*)d_in[8];
    const float* rmean = (const float*)d_in[9];
    const float* rvar  = (const float*)d_in[10];

    char* ws = (char*)d_ws;
    _Float16* xt16   = (_Float16*)(ws);                    //  8,388,608 B
    _Float16* Wv16   = (_Float16*)(ws +  8388608);         //    131,072
    _Float16* Wom16  = (_Float16*)(ws +  8519680);         //     65,536
    _Float16* Wo16   = (_Float16*)(ws +  8585216);         //    131,072
    float*    v32    = (float*)   (ws +  8716288);         // 16,777,216
    float*    om32   = (float*)   (ws + 25493504);         //  7,340,032
    _Float16* samp16 = (_Float16*)(ws + 32833536);         //  8,388,608

    convert_x_kernel<<<4096, 256, 0, stream>>>(x, xt16);
    convert_w_kernel<<<256, 256, 0, stream>>>(Wv,  Wv16,  256, 256);
    convert_w_kernel<<<128, 256, 0, stream>>>(Wom, Wom16, 112, 128);
    convert_w_kernel<<<256, 256, 0, stream>>>(Wo,  Wo16,  256, 256);

    // v = xt @ Wv^T + bv     (M=16384, N=256)
    gemm_wmma_kernel<<<dim3(MTOT / BM, 4), 256, 0, stream>>>(
        xt16, Wv16, bv, v32, 256, 256, 0, nullptr, nullptr, nullptr, nullptr);

    // om = xt @ Wom^T + bom  (M=16384, N=112 padded 128)
    gemm_wmma_kernel<<<dim3(MTOT / BM, 2), 256, 0, stream>>>(
        xt16, Wom16, bom, om32, 112, 112, 0, nullptr, nullptr, nullptr, nullptr);

    // deformable sampling -> f16 [M][256]
    sample_kernel<<<4096, 256, 0, stream>>>(v32, om32, samp16);

    // y = samp @ Wo^T + bo, fused BN + SiLU, NCHW store
    gemm_wmma_kernel<<<dim3(MTOT / BM, 4), 256, 0, stream>>>(
        samp16, Wo16, bo, (float*)d_out, 256, 256, 1, gamma, beta, rmean, rvar);
}
